// ALL_MOE_59090160058986
// MI455X (gfx1250) — matrix operand order repaired and checked by reading, not compiled, not run on hardware
//
#include <hip/hip_runtime.h>
#include <stdint.h>

// ---------------- problem constants (from reference) ----------------
#define BS      2048          // B*T
#define D_DIM   768
#define H_HEADS 4
#define N_KEYS  25600
#define KD_DIM  128
#define KNN     16
#define HID_A   44
#define HID_S   1024
#define QDIM    512           // H*KD

// scores kernel tiling
#define SC_M    32            // tokens per workgroup
#define SC_N    64            // key columns per N-tile
#define SC_NT   (N_KEYS / SC_N)   // 400 tiles

// ---------------- WMMA types ----------------
typedef __attribute__((ext_vector_type(16))) __bf16 v16bf;
typedef __attribute__((ext_vector_type(8)))  float  v8f;
typedef __attribute__((ext_vector_type(4)))  int    v4i;
typedef __attribute__((address_space(1))) v4i g_v4i;   // global int4
typedef __attribute__((address_space(3))) v4i l_v4i;   // LDS int4

#define WMMA_BF16(A, B, C) \
  __builtin_amdgcn_wmma_f32_16x16x32_bf16(false, (A), false, (B), (short)0, (C), false, false)
#define VZERO {0.f,0.f,0.f,0.f,0.f,0.f,0.f,0.f}

// CDNA5 async global->LDS path (ASYNCcnt), guarded so compile never regresses
#if __has_builtin(__builtin_amdgcn_global_load_async_to_lds_b128) && \
    __has_builtin(__builtin_amdgcn_s_wait_asynccnt)
#define USE_ASYNC_LDS 1
#else
#define USE_ASYNC_LDS 0
#endif

union FragU { uint4 u[2]; v16bf v; };

// A-matrix 16x32 bf16 fragment (ISA 7.12.2): lane L: M=L%16,
// element e -> K = (e<8?0:16) + (L>=16?8:0) + (e&7)  => two 16B chunks
__device__ __forceinline__ v16bf frag_a(const uint16_t* base, int ld, int m0, int k0, int lane) {
  int m   = m0 + (lane & 15);
  int off = (lane >> 4) << 3;           // 0 or 8 halves
  const uint16_t* p = base + (size_t)m * ld + k0 + off;
  FragU f;
  f.u[0] = *(const uint4*)(p);
  f.u[1] = *(const uint4*)(p + 16);
  return f.v;
}

// B-matrix 32x16 bf16 fragment, NT layout: lane L: N=L%16,
// element e -> K = (e<8?0:16) + (L>=16?8:0) + (e&7)  => two 16B chunks
__device__ __forceinline__ v16bf frag_b(const uint16_t* base, int ld, int n0, int k0, int lane) {
  int n   = n0 + (lane & 15);
  int off = (lane >> 4) << 3;           // 0 or 8 halves
  const uint16_t* p = base + (size_t)n * ld + k0 + off;
  FragU f;
  f.u[0] = *(const uint4*)(p);
  f.u[1] = *(const uint4*)(p + 16);
  return f.v;
}

__device__ __forceinline__ uint16_t f2bf(float x) {   // round-to-nearest-even
  unsigned b = __float_as_uint(x);
  return (uint16_t)((b + 0x7FFFu + ((b >> 16) & 1u)) >> 16);
}

// ---------------- conversion ----------------
__global__ void k_f32_to_bf16(const float* __restrict__ in, uint16_t* __restrict__ out, int n) {
  for (int i = blockIdx.x * blockDim.x + threadIdx.x; i < n; i += gridDim.x * blockDim.x)
    out[i] = f2bf(in[i]);
}

// per-token sum over D, zero comb accumulator
__global__ void k_xsum(const float* __restrict__ x, float* __restrict__ xsum,
                       float* __restrict__ comb) {
  int lane = threadIdx.x & 31, w = threadIdx.x >> 5;
  int t = blockIdx.x * 8 + w;
  float s = 0.f;
  for (int k = lane; k < D_DIM; k += 32) s += x[(size_t)t * D_DIM + k];
  for (int d = 16; d; d >>= 1) s += __shfl_xor(s, d);
  if (lane == 0) { xsum[t] = s; comb[t] = 0.f; }
}

// ---------------- q projection + BatchNorm (inference) ----------------
// two M-tiles per wave share one B fragment (halves wq L2 traffic)
__global__ void __launch_bounds__(128)
k_proj_bn(const uint16_t* __restrict__ xbf, const uint16_t* __restrict__ wqbf,
          const float* __restrict__ bq,   const float* __restrict__ gamma,
          const float* __restrict__ beta, const float* __restrict__ mean,
          const float* __restrict__ var,  uint16_t* __restrict__ qbf) {
  int lane = threadIdx.x & 31, w = threadIdx.x >> 5;
  int m0 = blockIdx.x * 32;
  int n0 = (blockIdx.y * 4 + w) * 16;
  v8f c0 = VZERO, c1 = VZERO;
  for (int kc = 0; kc < D_DIM; kc += 32) {
    v16bf b  = frag_b(wqbf, D_DIM, n0, kc, lane);
    v16bf x0 = frag_a(xbf,  D_DIM, m0,      kc, lane);
    v16bf x1 = frag_a(xbf,  D_DIM, m0 + 16, kc, lane);
    c0 = WMMA_BF16(x0, b, c0);
    c1 = WMMA_BF16(x1, b, c1);
  }
  int   n     = n0 + (lane & 15);
  float scale = gamma[n] * rsqrtf(var[n] + 1e-5f);
  float add   = bq[n] - mean[n];
  float bet   = beta[n];
  int   mh    = m0 + ((lane >> 4) << 3);
  for (int r = 0; r < 8; ++r)
    qbf[(size_t)(mh + r) * QDIM + n] = f2bf((c0[r] + add) * scale + bet);
  for (int r = 0; r < 8; ++r)
    qbf[(size_t)(mh + 16 + r) * QDIM + n] = f2bf((c1[r] + add) * scale + bet);
}

// ---------------- fused scores GEMM + streaming top-16 + expert path ----------------
// workgroup = (32 tokens, 1 head), 8 waves; keys streamed in 64-wide N-tiles,
// double-buffered through LDS with async global->LDS copies when available.
__global__ void __launch_bounds__(256)
k_scores_topk(const uint16_t* __restrict__ qbf, const uint16_t* __restrict__ keysbf,
              const float* __restrict__ w_down, const float* __restrict__ w_up,
              const float* __restrict__ a_w1,   const float* __restrict__ a_w2,
              const float* __restrict__ a_w3,   const float* __restrict__ xsum,
              float* __restrict__ comb) {
  __shared__ float sc[SC_M * SC_N];                       // 8 KB score staging (reused later)
#if USE_ASYNC_LDS
  __shared__ __align__(16) uint16_t keybuf[2][SC_N * KD_DIM];  // 2 x 16 KB double buffer
#endif
  int lane = threadIdx.x & 31, w = threadIdx.x >> 5;
  int half = lane >> 4, s = lane & 15;
  int m0 = blockIdx.x * SC_M, h = blockIdx.y;
  const uint16_t* keyh = keysbf + (size_t)h * N_KEYS * KD_DIM;

  int rt = w >> 2;                 // row-tile (0/1) for this wave
  int cb = (w & 3) * 16;           // column base within the 64-wide tile

  // A fragments (q rows m0+rt*16, full KD=128) preloaded once
  v16bf a0 = frag_a(qbf, QDIM, m0 + rt * 16, h * KD_DIM +  0, lane);
  v16bf a1 = frag_a(qbf, QDIM, m0 + rt * 16, h * KD_DIM + 32, lane);
  v16bf a2 = frag_a(qbf, QDIM, m0 + rt * 16, h * KD_DIM + 64, lane);
  v16bf a3 = frag_a(qbf, QDIM, m0 + rt * 16, h * KD_DIM + 96, lane);

  int   hw = w * 2 + half;         // half-wave id 0..15; owns rows hw and hw+16
  float topv[2] = {-INFINITY, -INFINITY};
  int   topi[2] = {0, 0};
  float thr [2] = {-INFINITY, -INFINITY};

#if USE_ASYNC_LDS
  auto issue_copy = [&](int tile, int buf) {
    // one 64x128-half key tile is a contiguous 16 KB block: 1024 b128 chunks
    const uint16_t* src = keyh + (size_t)tile * SC_N * KD_DIM;
    for (int i = 0; i < 4; ++i) {
      int c = threadIdx.x + i * 256;
      __builtin_amdgcn_global_load_async_to_lds_b128(
          (g_v4i*)(src + (size_t)c * 8),
          (l_v4i*)(&keybuf[buf][c * 8]), 0, 0);
    }
  };
  issue_copy(0, 0);
#endif

  for (int t = 0; t < SC_NT; ++t) {
#if USE_ASYNC_LDS
    issue_copy((t + 1) % SC_NT, (t + 1) & 1);       // prefetch next tile
    __builtin_amdgcn_s_wait_asynccnt(4);            // oldest 4 (current tile) landed
    __syncthreads();                                // all waves' copies + prev merge done
    const uint16_t* kb = &keybuf[t & 1][0];
    const int co = 0;
#else
    if (t + 1 < SC_NT)
      __builtin_prefetch(keyh + (size_t)(t + 1) * SC_N * KD_DIM, 0, 1);
    __syncthreads();                                // prev merge finished reading sc
    const uint16_t* kb = keyh;
    const int co = t * SC_N;
#endif
    v8f c = VZERO;
    v16bf b;
    b = frag_b(kb, KD_DIM, co + cb,  0, lane); c = WMMA_BF16(a0, b, c);
    b = frag_b(kb, KD_DIM, co + cb, 32, lane); c = WMMA_BF16(a1, b, c);
    b = frag_b(kb, KD_DIM, co + cb, 64, lane); c = WMMA_BF16(a2, b, c);
    b = frag_b(kb, KD_DIM, co + cb, 96, lane); c = WMMA_BF16(a3, b, c);

    // stage 32x64 scores
    {
      int rr0 = rt * 16 + (half << 3);
      for (int r = 0; r < 8; ++r) sc[(rr0 + r) * SC_N + cb + s] = c[r];
    }
    __syncthreads();

    // threshold-filtered top-16 merge: half-wave owns rows hw, hw+16; 4 cols/sublane
    for (int rr = 0; rr < 2; ++rr) {
      int row = hw + rr * 16;
      for (int j = 0; j < 4; ++j) {
        float v = sc[row * SC_N + j * 16 + s];
        unsigned long long bal = __ballot(v > thr[rr]);
        unsigned bits = (unsigned)(bal >> (half * 16)) & 0xFFFFu;
        while (bits) {                  // rare (~118 insertions / row over all N)
          int src = __ffs(bits) - 1;
          bits &= bits - 1;
          float cv = __shfl(v, src, 16);
          int   cn = t * SC_N + j * 16 + src;
          float mv = topv[rr]; int ml = s;
          for (int d = 1; d < 16; d <<= 1) {
            float ov = __shfl_xor(mv, d, 16);
            int   ol = __shfl_xor(ml, d, 16);
            if (ov < mv || (ov == mv && ol < ml)) { mv = ov; ml = ol; }
          }
          if (cv > mv && s == ml) { topv[rr] = cv; topi[rr] = cn; }
          float t2 = topv[rr];
          for (int d = 1; d < 16; d <<= 1) t2 = fminf(t2, __shfl_xor(t2, d, 16));
          thr[rr] = t2;
        }
      }
    }
  }
  __syncthreads();

  // per-row softmax + expert-dim-1 gather + tiny SwiGLU (16 -> 44 -> 16)
  float* xdl  = sc;                  // 32 x 16
  float* hidl = sc + SC_M * 16;      // 32 x 44  (total 1920 floats <= 2048)
  float gate[2], xd_r[2], wu_r[2];
  for (int rr = 0; rr < 2; ++rr) {
    float mx = topv[rr];
    for (int d = 1; d < 16; d <<= 1) mx = fmaxf(mx, __shfl_xor(mx, d, 16));
    float e = __expf(topv[rr] - mx);
    float es = e;
    for (int d = 1; d < 16; d <<= 1) es += __shfl_xor(es, d, 16);
    gate[rr] = e / es;
    int token = m0 + hw + rr * 16;
    xd_r[rr] = xsum[token] * w_down[topi[rr]];
    wu_r[rr] = w_up[topi[rr]];
    xdl[(hw + rr * 16) * 16 + s] = xd_r[rr];
  }
  __syncthreads();
  for (int rr = 0; rr < 2; ++rr) {
    int row = hw + rr * 16;
    for (int tj = 0; tj < 3; ++tj) {
      int j = s + 16 * tj;
      if (j < HID_A) {
        float z1 = 0.f, z3 = 0.f;
        for (int k = 0; k < 16; ++k) {
          float xv = xdl[row * 16 + k];
          z1 += a_w1[j * 16 + k] * xv;
          z3 += a_w3[j * 16 + k] * xv;
        }
        float sig = 1.f / (1.f + __expf(-z1));
        hidl[row * HID_A + j] = (z1 * sig) * z3;
      }
    }
  }
  __syncthreads();
  for (int rr = 0; rr < 2; ++rr) {
    int row = hw + rr * 16;
    float hk = 0.f;
    for (int j = 0; j < HID_A; ++j) hk += a_w2[s * HID_A + j] * hidl[row * HID_A + j];
    float contrib = hk * gate[rr] * wu_r[rr];
    for (int d = 1; d < 16; d <<= 1) contrib += __shfl_xor(contrib, d, 16);
    if (s == 0) atomicAdd(&comb[m0 + row], contrib);
  }
}

// ---------------- dense SwiGLU: mid = silu(x@w1^T) * (x@w3^T), bf16 ----------------
// two M-tiles per wave; shared A loads, shared B fragments across accumulators
__global__ void __launch_bounds__(128)
k_swiglu_mid(const uint16_t* __restrict__ xbf, const uint16_t* __restrict__ w1bf,
             const uint16_t* __restrict__ w3bf, uint16_t* __restrict__ midbf) {
  int lane = threadIdx.x & 31, w = threadIdx.x >> 5;
  int m0 = blockIdx.x * 32;
  int n0 = (blockIdx.y * 4 + w) * 16;
  v8f c10 = VZERO, c30 = VZERO, c11 = VZERO, c31 = VZERO;
  for (int kc = 0; kc < D_DIM; kc += 32) {
    v16bf b1 = frag_b(w1bf, D_DIM, n0, kc, lane);
    v16bf b3 = frag_b(w3bf, D_DIM, n0, kc, lane);
    v16bf x0 = frag_a(xbf,  D_DIM, m0,      kc, lane);
    v16bf x1 = frag_a(xbf,  D_DIM, m0 + 16, kc, lane);
    c10 = WMMA_BF16(x0, b1, c10);
    c30 = WMMA_BF16(x0, b3, c30);
    c11 = WMMA_BF16(x1, b1, c11);
    c31 = WMMA_BF16(x1, b3, c31);
  }
  int n  = n0 + (lane & 15);
  int mh = m0 + ((lane >> 4) << 3);
  for (int r = 0; r < 8; ++r) {
    float z1 = c10[r], z3 = c30[r];
    midbf[(size_t)(mh + r) * HID_S + n] = f2bf(z1 / (1.f + __expf(-z1)) * z3);
  }
  for (int r = 0; r < 8; ++r) {
    float z1 = c11[r], z3 = c31[r];
    midbf[(size_t)(mh + 16 + r) * HID_S + n] = f2bf(z1 / (1.f + __expf(-z1)) * z3);
  }
}

// ---------------- out = mid @ w2^T + comb ----------------
__global__ void __launch_bounds__(128)
k_out(const uint16_t* __restrict__ midbf, const uint16_t* __restrict__ w2bf,
      const float* __restrict__ comb, float* __restrict__ out) {
  int lane = threadIdx.x & 31, w = threadIdx.x >> 5;
  int m0 = blockIdx.x * 32;
  int n0 = (blockIdx.y * 4 + w) * 16;
  v8f c0 = VZERO, c1 = VZERO;
  for (int kc = 0; kc < HID_S; kc += 32) {
    v16bf b  = frag_b(w2bf,  HID_S, n0, kc, lane);
    v16bf a0 = frag_a(midbf, HID_S, m0,      kc, lane);
    v16bf a1 = frag_a(midbf, HID_S, m0 + 16, kc, lane);
    c0 = WMMA_BF16(a0, b, c0);
    c1 = WMMA_BF16(a1, b, c1);
  }
  int n  = n0 + (lane & 15);
  int mh = m0 + ((lane >> 4) << 3);
  for (int r = 0; r < 8; ++r)
    out[(size_t)(mh + r) * D_DIM + n] = c0[r] + comb[mh + r];
  for (int r = 0; r < 8; ++r)
    out[(size_t)(mh + 16 + r) * D_DIM + n] = c1[r] + comb[mh + 16 + r];
}

// ---------------- host launcher ----------------
extern "C" void kernel_launch(void* const* d_in, const int* in_sizes, int n_in,
                              void* d_out, int out_size, void* d_ws, size_t ws_size,
                              hipStream_t stream) {
  const float* x        = (const float*)d_in[0];
  const float* wq       = (const float*)d_in[1];
  const float* bq       = (const float*)d_in[2];
  const float* bn_gamma = (const float*)d_in[3];
  const float* bn_beta  = (const float*)d_in[4];
  const float* bn_mean  = (const float*)d_in[5];
  const float* bn_var   = (const float*)d_in[6];
  const float* keys     = (const float*)d_in[7];
  const float* w_down   = (const float*)d_in[8];
  const float* w_up     = (const float*)d_in[9];
  const float* a_w1     = (const float*)d_in[10];
  const float* a_w2     = (const float*)d_in[11];
  const float* a_w3     = (const float*)d_in[12];
  const float* s_w1     = (const float*)d_in[13];
  const float* s_w2     = (const float*)d_in[14];
  const float* s_w3     = (const float*)d_in[15];
  float* out = (float*)d_out;

  const size_t nX   = (size_t)BS * D_DIM;
  const size_t nWQ  = (size_t)QDIM * D_DIM;
  const size_t nKEY = (size_t)H_HEADS * N_KEYS * KD_DIM;
  const size_t nSW  = (size_t)HID_S * D_DIM;
  const size_t nQ   = (size_t)BS * QDIM;
  const size_t nMID = (size_t)BS * HID_S;

  uint16_t* x_bf   = (uint16_t*)d_ws;
  uint16_t* wq_bf  = x_bf   + nX;
  uint16_t* key_bf = wq_bf  + nWQ;
  uint16_t* w1_bf  = key_bf + nKEY;
  uint16_t* w3_bf  = w1_bf  + nSW;
  uint16_t* w2_bf  = w3_bf  + nSW;
  uint16_t* q_bf   = w2_bf  + nSW;
  uint16_t* mid_bf = q_bf   + nQ;
  float*    xsum   = (float*)(mid_bf + nMID);
  float*    comb   = xsum + BS;

  auto cvt = [&](const float* src, uint16_t* dst, size_t n) {
    int blocks = (int)((n + 255) / 256);
    if (blocks > 4096) blocks = 4096;
    k_f32_to_bf16<<<blocks, 256, 0, stream>>>(src, dst, (int)n);
  };
  cvt(x,    x_bf,   nX);
  cvt(wq,   wq_bf,  nWQ);
  cvt(keys, key_bf, nKEY);
  cvt(s_w1, w1_bf,  nSW);
  cvt(s_w3, w3_bf,  nSW);
  cvt(s_w2, w2_bf,  nSW);

  k_xsum<<<BS / 8, 256, 0, stream>>>(x, xsum, comb);

  k_proj_bn<<<dim3(BS / 32, QDIM / 64), 128, 0, stream>>>(
      x_bf, wq_bf, bq, bn_gamma, bn_beta, bn_mean, bn_var, q_bf);

  k_scores_topk<<<dim3(BS / SC_M, H_HEADS), 256, 0, stream>>>(
      q_bf, key_bf, w_down, w_up, a_w1, a_w2, a_w3, xsum, comb);

  k_swiglu_mid<<<dim3(BS / 32, HID_S / 64), 128, 0, stream>>>(x_bf, w1_bf, w3_bf, mid_bf);
  k_out<<<dim3(BS / 32, D_DIM / 64), 128, 0, stream>>>(mid_bf, w2_bf, comb, out);

  (void)in_sizes; (void)n_in; (void)out_size; (void)ws_size;
}
